// GroupQueryAttention_88897233093062
// MI455X (gfx1250) — compile-verified, matrix operand order checked by reading, not run on hardware
//
#include <hip/hip_runtime.h>
#include <hip/hip_bf16.h>
#include <cstdint>

typedef __attribute__((ext_vector_type(16))) _Float16 v16h;
typedef __attribute__((ext_vector_type(8)))  _Float16 v8h;
typedef __attribute__((ext_vector_type(4)))  _Float16 v4h;
typedef __attribute__((ext_vector_type(8)))  float    v8f;

#define E_    1024
#define H_    16
#define DH_   64
#define G_    8
#define N_    8
#define L_    4096
#define C_    (L_ / G_)      // 512 sequence chunks
#define XPAD  1032           // padded f16 row stride for staged X tiles (bank-spread)
#define PPAD  72             // padded f16 row stride for per-head projection tiles

// ---------------------------------------------------------------------------
// WMMA fragment loader. CDNA5 ISA layout for 16-bit A (16x32) / B (32x16):
// lane m = lane&15, half = lane>>4; elements 0..7 hold K = half*8 + 0..7,
// elements 8..15 hold K = 16 + half*8 + 0..7. Two 16-byte loads per lane.
// ---------------------------------------------------------------------------
__device__ __forceinline__ v16h load_frag(const _Float16* p, int stride, int row0, int k0) {
    const int lane = threadIdx.x & 31;
    const int m    = lane & 15;
    const int half = lane >> 4;
    const _Float16* r = p + (size_t)(row0 + m) * stride + k0 + half * 8;
    v8h lo = *(const v8h*)(r);        // K = k0 + half*8 + 0..7
    v8h hi = *(const v8h*)(r + 16);   // K = k0 + 16 + half*8 + 0..7
    v16h out;
#pragma unroll
    for (int e = 0; e < 8; ++e) { out[e] = lo[e]; out[e + 8] = hi[e]; }
    return out;
}

__device__ __forceinline__ v8f wmma_f16(v16h a, v16h b, v8f c) {
    return __builtin_amdgcn_wmma_f32_16x16x32_f16(false, a, false, b, (short)0, c,
                                                  false, false);
}

// ---------------------------------------------------------------------------
// One-time weight conversion f32 -> f16 into workspace (Wq|Wk|Wv|Wo contiguous)
// ---------------------------------------------------------------------------
__global__ __launch_bounds__(256) void cvt_weights(const float* __restrict__ a,
                                                   const float* __restrict__ b,
                                                   const float* __restrict__ c,
                                                   const float* __restrict__ d,
                                                   _Float16* __restrict__ o) {
    const size_t idx = (size_t)blockIdx.x * 256 + threadIdx.x;   // 0 .. 4M-1
    const int which = (int)(idx >> 20);
    const int off   = (int)(idx & 1048575);
    const float* srcs[4] = {a, b, c, d};
    o[idx] = (_Float16)srcs[which][off];
}

// ---------------------------------------------------------------------------
// Fused Q/K projection + block-local attention weights.
// grid = C (one chunk per workgroup), 512 threads = 16 wave32.
// 64 rows (r = n*8 + g) x E of q,k staged in LDS as f16 (264 KB).
// Head loop processes TWO heads per iteration; per head 32 output tiles
// (2 matrices x 4 row-tiles x 4 col-tiles); each of the 16 waves owns a
// 2x2 tile block -> every A and every B fragment feeds 2 wmmas, cutting
// L2 weight-fragment traffic 4x vs naive and LDS traffic 2x.
// ---------------------------------------------------------------------------
__global__ __launch_bounds__(512)
void gqa_proj_attn(const float* __restrict__ q, const float* __restrict__ k,
                   const _Float16* __restrict__ Wq16, const _Float16* __restrict__ Wk16,
                   const float* __restrict__ bq, const float* __restrict__ bk,
                   float* __restrict__ attn_out, float* __restrict__ wlast) {
    __shared__ __align__(16) _Float16 Xq[64 * XPAD];        // 132,096 B
    __shared__ __align__(16) _Float16 Xk[64 * XPAD];        // 132,096 B
    __shared__ __align__(16) _Float16 QPh[2][64 * PPAD];    //  18,432 B
    __shared__ __align__(16) _Float16 KPh[2][64 * PPAD];    //  18,432 B
    __shared__ float sc[2 * 512];                           //   4,096 B
    __shared__ float wsum[2 * 512];                         //   4,096 B

    const int tid  = threadIdx.x;
    const int c    = blockIdx.x;
    const int wave = tid >> 5;
    const int lane = tid & 31;
    const bool lastChunk = (c == C_ - 1);

    wsum[tid]       = 0.0f;
    wsum[tid + 512] = 0.0f;

    // ---- stage q,k rows into LDS as f16 : row r = n*8 + g ----
    {
        const int r   = tid >> 3;            // 0..63
        const int n   = r >> 3;
        const int g   = r & 7;
        const int seq = c * G_ + g;
        const float4* gq = (const float4*)(q + (size_t)(seq * N_ + n) * E_);
        const float4* gk = (const float4*)(k + (size_t)(seq * N_ + n) * E_);
        const int c0 = (tid & 7) * 32;       // float4 units
#pragma unroll 4
        for (int it = 0; it < 32; ++it) {
            const float4 aq = gq[c0 + it];
            const float4 ak = gk[c0 + it];
            const int col = (c0 + it) * 4;
            v4h hq, hk;
            hq[0] = (_Float16)aq.x; hq[1] = (_Float16)aq.y;
            hq[2] = (_Float16)aq.z; hq[3] = (_Float16)aq.w;
            hk[0] = (_Float16)ak.x; hk[1] = (_Float16)ak.y;
            hk[2] = (_Float16)ak.z; hk[3] = (_Float16)ak.w;
            *(v4h*)(&Xq[r * XPAD + col]) = hq;
            *(v4h*)(&Xk[r * XPAD + col]) = hk;
        }
    }
    __syncthreads();

    // static per-wave tile decode: head slot, matrix, 2x2 tile block
    const int hsel = wave >> 3;          // which of the two heads this iteration
    const int w7   = wave & 7;
    const int msel = w7 >> 2;            // 0 = Q, 1 = K
    const int i2   = ((w7 >> 1) & 1) * 2;// row tiles i2, i2+1 (rows 16*i2 ..)
    const int j2   = (w7 & 1) * 2;       // col tiles j2, j2+1
    const _Float16* X    = msel ? Xk : Xq;
    const _Float16* W    = msel ? Wk16 : Wq16;
    const float*    bias = msel ? bk : bq;
    _Float16*       P    = msel ? &KPh[hsel][0] : &QPh[hsel][0];

    for (int hh = 0; hh < H_ / 2; ++hh) {
        const int h       = hh * 2 + hsel;
        const int colbase = h * DH_;

        // ---- P1: projection, 2x2 tile block per wave ----
        {
            const _Float16* wrow = W + (size_t)(colbase + 16 * j2 + (lane & 15)) * E_
                                     + (lane >> 4) * 8;
            v8f acc00 = {}, acc01 = {}, acc10 = {}, acc11 = {};
            for (int kk = 0; kk < 32; ++kk) {
                if (kk + 8 < 32) __builtin_prefetch(wrow + (kk + 8) * 32, 0, 1);
                const v16h a0 = load_frag(X, XPAD, 16 * i2, kk * 32);
                const v16h a1 = load_frag(X, XPAD, 16 * i2 + 16, kk * 32);
                const v16h b0 = load_frag(W, E_, colbase + 16 * j2, kk * 32);
                const v16h b1 = load_frag(W, E_, colbase + 16 * j2 + 16, kk * 32);
                acc00 = wmma_f16(a0, b0, acc00);
                acc01 = wmma_f16(a0, b1, acc01);
                acc10 = wmma_f16(a1, b0, acc10);
                acc11 = wmma_f16(a1, b1, acc11);
            }
            const int half = lane >> 4;
            const int cl0  = colbase + 16 * j2 + (lane & 15);
            const float bb0 = bias[cl0];
            const float bb1 = bias[cl0 + 16];
            const v8f* accs[2][2] = {{&acc00, &acc01}, {&acc10, &acc11}};
#pragma unroll
            for (int ii = 0; ii < 2; ++ii) {
#pragma unroll
                for (int jj = 0; jj < 2; ++jj) {
                    const v8f acc = *accs[ii][jj];
                    const float bb = jj ? bb1 : bb0;
#pragma unroll
                    for (int vv = 0; vv < 8; ++vv) {
                        const int row = 16 * (i2 + ii) + vv + 8 * half;
                        P[row * PPAD + 16 * (j2 + jj) + (lane & 15)] =
                            (_Float16)(acc[vv] + bb);
                    }
                }
            }
        }
        __syncthreads();

        // ---- P2: scores. 8 tiles = (head slot, n-pair p); K = Dh = 64 ----
        if (wave < 8) {
            const int hs = wave >> 2;
            const int p  = wave & 3;
            v8f acc = {};
#pragma unroll
            for (int kk = 0; kk < 2; ++kk) {
                const v16h a = load_frag(&QPh[hs][0], PPAD, 16 * p, kk * 32);
                const v16h b = load_frag(&KPh[hs][0], PPAD, 16 * p, kk * 32);
                acc = wmma_f16(a, b, acc);
            }
            const int half = lane >> 4;
            if (half == ((lane >> 3) & 1)) {   // block-diagonal (same n) entries
                const int n  = 2 * p + half;
                const int gk = lane & 7;
#pragma unroll
                for (int vv = 0; vv < 8; ++vv)
                    sc[hs * 512 + (n * 8 + vv) * 8 + gk] = acc[vv] * 0.125f;
            }
        }
        __syncthreads();

        // ---- softmax over s, accumulate per-head-slot mean banks ----
        if (tid < 128) {
            const int hs = tid >> 6;
            const int r  = tid & 63;
            const int n  = r >> 3;
            const int gq = r & 7;
            const int hcur = hh * 2 + hs;
            float srow[8];
            float mx = -3.0e38f;
#pragma unroll
            for (int s = 0; s < 8; ++s) {
                srow[s] = sc[hs * 512 + (n * 8 + gq) * 8 + s];
                mx = fmaxf(mx, srow[s]);
            }
            float sum = 0.0f;
#pragma unroll
            for (int s = 0; s < 8; ++s) { srow[s] = __expf(srow[s] - mx); sum += srow[s]; }
            const float inv = 1.0f / sum;
#pragma unroll
            for (int s = 0; s < 8; ++s) {
                const float w = srow[s] * inv;
                wsum[hs * 512 + (n * 8 + gq) * 8 + s] += w;
                if (lastChunk)
                    wlast[(((n * H_ + hcur) * G_ + gq) * G_) + s] = w;
            }
        }
        __syncthreads();
    }

    // ---- write head-averaged attention weights: attn[(c*N+n)*64 + g*8 + s] ----
    {
        const int n    = tid >> 6;        // 0..7
        const int rest = tid & 63;
        attn_out[(size_t)(c * N_ + n) * 64 + rest] =
            (wsum[tid] + wsum[512 + tid]) * (1.0f / 16.0f);
    }
}

// ---------------------------------------------------------------------------
// Last-chunk V projection + attention combine -> OUT (f16, rows g*8+n) in ws.
// Single workgroup (tiny fraction of total work). 1x2 tile blocks per wave.
// ---------------------------------------------------------------------------
__global__ __launch_bounds__(256)
void gqa_vproj_combine(const float* __restrict__ vsrc, const _Float16* __restrict__ Wv16,
                       const float* __restrict__ bv, const float* __restrict__ wlast,
                       _Float16* __restrict__ outh) {
    __shared__ __align__(16) _Float16 Xv[64 * XPAD];   // 132,096 B
    __shared__ __align__(16) _Float16 VPh[64 * PPAD];  //   9,216 B
    const int tid  = threadIdx.x;
    const int wave = tid >> 5;
    const int lane = tid & 31;

    // stage last-chunk v rows: row r = n*8 + s
    {
        const int r   = tid >> 2;         // 0..63
        const int n   = r >> 3;
        const int s   = r & 7;
        const int seq = (C_ - 1) * G_ + s;
        const float4* gv = (const float4*)(vsrc + (size_t)(seq * N_ + n) * E_);
        const int c0 = (tid & 3) * 64;    // float4 units
        for (int it = 0; it < 64; ++it) {
            const float4 a = gv[c0 + it];
            const int col = (c0 + it) * 4;
            v4h hv;
            hv[0] = (_Float16)a.x; hv[1] = (_Float16)a.y;
            hv[2] = (_Float16)a.z; hv[3] = (_Float16)a.w;
            *(v4h*)(&Xv[r * XPAD + col]) = hv;
        }
    }
    __syncthreads();

    const int ti = wave >> 1;           // row tile 0..3
    const int j2 = (wave & 1) * 2;      // col tiles j2, j2+1

    for (int h = 0; h < H_; ++h) {
        const int colbase = h * DH_;
        // VP_h = Xv @ Wv_h^T + bv : 4x4 tiles, 1x2 block per wave
        {
            v8f acc0 = {}, acc1 = {};
            for (int kk = 0; kk < 32; ++kk) {
                const v16h a  = load_frag(Xv, XPAD, 16 * ti, kk * 32);
                const v16h b0 = load_frag(Wv16, E_, colbase + 16 * j2, kk * 32);
                const v16h b1 = load_frag(Wv16, E_, colbase + 16 * j2 + 16, kk * 32);
                acc0 = wmma_f16(a, b0, acc0);
                acc1 = wmma_f16(a, b1, acc1);
            }
            const int half = lane >> 4;
            const int cl0  = colbase + 16 * j2 + (lane & 15);
            const float bb0 = bv[cl0];
            const float bb1 = bv[cl0 + 16];
#pragma unroll
            for (int vv = 0; vv < 8; ++vv) {
                const int row = 16 * ti + vv + 8 * half;
                VPh[row * PPAD + 16 * j2 + (lane & 15)]      = (_Float16)(acc0[vv] + bb0);
                VPh[row * PPAD + 16 * j2 + 16 + (lane & 15)] = (_Float16)(acc1[vv] + bb1);
            }
        }
        __syncthreads();

        // combine: out[g,n, colbase+dd] = sum_s w[n,h,g,s] * VP_h[n*8+s][dd]
        {
            const int e0  = tid * 16;
            const int g   = e0 >> 9;
            const int n   = (e0 >> 6) & 7;
            const int dd0 = e0 & 63;
            float w[8];
#pragma unroll
            for (int s = 0; s < 8; ++s) w[s] = wlast[(((n * H_ + h) * G_ + g) * G_) + s];
#pragma unroll
            for (int ii = 0; ii < 16; ++ii) {
                const int dd = dd0 + ii;
                float sum = 0.0f;
#pragma unroll
                for (int s = 0; s < 8; ++s)
                    sum += w[s] * (float)VPh[(n * 8 + s) * PPAD + dd];
                outh[(size_t)(g * 8 + n) * E_ + colbase + dd] = (_Float16)sum;
            }
        }
        __syncthreads();
    }
}

// ---------------------------------------------------------------------------
// Final projection: x = OUT @ Wo^T + bo  -> d_out[0 : 64*1024]
// grid = 8 blocks x 8 waves = 64 waves; 256 tiles, 1x4 block per wave so the
// A fragment (global f16) is reused across 4 wmmas.
// ---------------------------------------------------------------------------
__global__ __launch_bounds__(256)
void gqa_out_proj(const _Float16* __restrict__ outh, const _Float16* __restrict__ Wo16,
                  const float* __restrict__ bo, float* __restrict__ x) {
    const int tid  = threadIdx.x;
    const int lane = tid & 31;
    const int gw   = blockIdx.x * 8 + (tid >> 5);   // 0..63
    const int ti   = gw >> 4;                        // row tile 0..3
    const int j0   = (gw & 15) * 4;                  // col tiles j0..j0+3

    v8f acc0 = {}, acc1 = {}, acc2 = {}, acc3 = {};
    for (int kk = 0; kk < 32; ++kk) {
        const v16h a  = load_frag(outh, E_, 16 * ti, kk * 32);
        const v16h b0 = load_frag(Wo16, E_, (j0 + 0) * 16, kk * 32);
        const v16h b1 = load_frag(Wo16, E_, (j0 + 1) * 16, kk * 32);
        const v16h b2 = load_frag(Wo16, E_, (j0 + 2) * 16, kk * 32);
        const v16h b3 = load_frag(Wo16, E_, (j0 + 3) * 16, kk * 32);
        acc0 = wmma_f16(a, b0, acc0);
        acc1 = wmma_f16(a, b1, acc1);
        acc2 = wmma_f16(a, b2, acc2);
        acc3 = wmma_f16(a, b3, acc3);
    }
    const int half = lane >> 4;
#pragma unroll
    for (int jj = 0; jj < 4; ++jj) {
        const v8f* accs[4] = {&acc0, &acc1, &acc2, &acc3};
        const v8f  acc     = *accs[jj];
        const int  col     = (j0 + jj) * 16 + (lane & 15);
        const float bb     = bo[col];
#pragma unroll
        for (int vv = 0; vv < 8; ++vv) {
            const int row = 16 * ti + vv + 8 * half;   // row = g*8 + n
            x[(size_t)row * E_ + col] = acc[vv] + bb;
        }
    }
}

// ---------------------------------------------------------------------------
extern "C" void kernel_launch(void* const* d_in, const int* in_sizes, int n_in,
                              void* d_out, int out_size, void* d_ws, size_t ws_size,
                              hipStream_t stream) {
    (void)in_sizes; (void)n_in; (void)out_size; (void)ws_size;
    const float* q  = (const float*)d_in[0];
    const float* k  = (const float*)d_in[1];
    const float* v  = (const float*)d_in[2];
    const float* Wq = (const float*)d_in[3];
    const float* Wk = (const float*)d_in[4];
    const float* Wv = (const float*)d_in[5];
    const float* bq = (const float*)d_in[6];
    const float* bk = (const float*)d_in[7];
    const float* bv = (const float*)d_in[8];
    const float* Wo = (const float*)d_in[9];
    const float* bo = (const float*)d_in[10];

    float* out      = (float*)d_out;
    float* x_out    = out;              // 64*1024
    float* attn_out = out + 64 * 1024;  // 512*8*64

    char* ws = (char*)d_ws;
    _Float16* W16   = (_Float16*)ws;                          // 4 x 1M halves (8 MB)
    _Float16* Wq16  = W16 + (size_t)0 * 1048576;
    _Float16* Wk16  = W16 + (size_t)1 * 1048576;
    _Float16* Wv16  = W16 + (size_t)2 * 1048576;
    _Float16* Wo16  = W16 + (size_t)3 * 1048576;
    float*    wlast = (float*)(ws + (size_t)4 * 2097152);             // 8192 f32
    _Float16* outh  = (_Float16*)(ws + (size_t)4 * 2097152 + 32768);  // 64*1024 f16

    cvt_weights<<<16384, 256, 0, stream>>>(Wq, Wk, Wv, Wo, W16);
    gqa_proj_attn<<<C_, 512, 0, stream>>>(q, k, Wq16, Wk16, bq, bk, attn_out, wlast);
    gqa_vproj_combine<<<1, 256, 0, stream>>>(v, Wv16, bv, wlast, outh);
    gqa_out_proj<<<8, 256, 0, stream>>>(outh, Wo16, bo, x_out);
}
